// BinMatmulAttention_22265110462923
// MI455X (gfx1250) — matrix operand order repaired and checked by reading, not compiled, not run on hardware
//
#include <hip/hip_runtime.h>
#include <hip/hip_bf16.h>

// ---- CDNA5 vector types (avoid HIP class-type vectors inside bit-casts) ----
typedef __attribute__((ext_vector_type(16))) __bf16        v16bf;
typedef __attribute__((ext_vector_type(4)))  __bf16        v4bf;
typedef __attribute__((ext_vector_type(8)))  float         v8f;
typedef __attribute__((ext_vector_type(4)))  float         v4f;
typedef __attribute__((ext_vector_type(4)))  unsigned int  v4u;

#define DHEAD   128
#define SEQ     2048
#define QTILE   128      // query rows per workgroup
#define KTILE   64       // kv rows per iteration
#define NWAVE   8
#define KT_STRIDE 136    // halves: 128 + 16B pad (16B-aligned rows, de-conflicted)
#define VT_STRIDE 72     // halves: 64 + 16B pad
#define PW_STRIDE 72     // halves: 64 + 16B pad

struct Pair16 { v4u a, b; };           // 32 bytes, bit-castable to v16bf

// Native fptrunc; backend picks the best gfx1250 lowering (packed cvt if present).
__device__ __forceinline__ unsigned short f2bf(float f) {
    return __builtin_bit_cast(unsigned short, (__bf16)f);
}

// Two 16B LDS loads assembled into one 32B bf16 fragment.
__device__ __forceinline__ v16bf ldsfrag(const unsigned short* base, int off0, int off1) {
    Pair16 p;
    p.a = *(const v4u*)(base + off0);
    p.b = *(const v4u*)(base + off1);
    return __builtin_bit_cast(v16bf, p);
}

// Convert 8 + 8 fp32 from global into one bf16 A-fragment (vector fptrunc).
__device__ __forceinline__ v16bf cvtfrag(const float* p0, const float* p1) {
    v16bf r;
#pragma unroll
    for (int i = 0; i < 8; ++i) r[i]     = (__bf16)p0[i];
#pragma unroll
    for (int i = 0; i < 8; ++i) r[8 + i] = (__bf16)p1[i];
    return r;
}

__launch_bounds__(256)
__global__ void fa_fwd_bf16wmma(const float* __restrict__ Q,
                                const float* __restrict__ K,
                                const float* __restrict__ V,
                                float* __restrict__ Out) {
    __shared__ __align__(16) unsigned short sK [KTILE * KT_STRIDE];          // K tile, bf16 row-major
    __shared__ __align__(16) unsigned short sVt[DHEAD * VT_STRIDE];          // V tile, bf16 TRANSPOSED [d][j]
    __shared__ __align__(16) unsigned short sP [NWAVE * 16 * PW_STRIDE];     // per-wave P scratch

    const int bh   = blockIdx.y;
    const int qb   = blockIdx.x * QTILE;
    const float* Qp = Q   + (size_t)bh * SEQ * DHEAD;
    const float* Kp = K   + (size_t)bh * SEQ * DHEAD;
    const float* Vp = V   + (size_t)bh * SEQ * DHEAD;
    float*       Op = Out + (size_t)bh * SEQ * DHEAD;

    const int tid  = threadIdx.x;
    const int wave = tid >> 5;
    const int lane = tid & 31;
    const int ln16 = lane & 15;
    const int hi   = lane >> 4;          // half-wave: 0 or 1

    // ---- Preload this wave's 16x128 Q tile as 4 bf16 A-fragments (K=32 each) ----
    // A layout: lanes 0-15 hold row M=lane, k-halves {0..7},{16..23}; lanes 16-31 row M=lane-16, {8..15},{24..31}
    const float* qrow = Qp + (size_t)(qb + wave * 16 + ln16) * DHEAD;
    v16bf qf[4];
#pragma unroll
    for (int kc = 0; kc < 4; ++kc) {
        const int lo = kc * 32 + hi * 8;
        qf[kc] = cvtfrag(qrow + lo, qrow + lo + 16);
    }

    // ---- Online-softmax state: rows r+8*hi for r=0..7 (matches f32 C/D layout) ----
    float m[8], l[8];
    v8f acc[8];
    const v8f vzero = {};
#pragma unroll
    for (int r = 0; r < 8; ++r) { m[r] = -__builtin_inff(); l[r] = 0.0f; acc[r] = vzero; }

    for (int kt = 0; kt < SEQ / KTILE; ++kt) {
        const int kb = kt * KTILE;
        __syncthreads();   // protect sK/sVt from previous iteration's readers

        // ---- Cooperative load: K tile -> bf16 LDS (row-major) ----
#pragma unroll
        for (int i = 0; i < 8; ++i) {
            const int idx = tid + i * 256;          // 0..2047 float4 slots
            const int row = idx >> 5;               // 32 float4 per 128-f32 row
            const int c4  = (idx & 31) * 4;
            v4f k4 = *(const v4f*)(Kp + (size_t)(kb + row) * DHEAD + c4);
            v4bf pk;
            pk[0] = (__bf16)k4.x; pk[1] = (__bf16)k4.y;
            pk[2] = (__bf16)k4.z; pk[3] = (__bf16)k4.w;
            *(v4bf*)(sK + row * KT_STRIDE + c4) = pk;   // 8B aligned (c4 % 4 == 0)
        }
        // ---- Cooperative load: V tile -> bf16 LDS, transposed to [d][j] ----
#pragma unroll
        for (int i = 0; i < 8; ++i) {
            const int idx = tid + i * 256;
            const int row = idx >> 5;
            const int c4  = (idx & 31) * 4;
            v4f vv = *(const v4f*)(Vp + (size_t)(kb + row) * DHEAD + c4);
            sVt[(c4 + 0) * VT_STRIDE + row] = f2bf(vv.x);
            sVt[(c4 + 1) * VT_STRIDE + row] = f2bf(vv.y);
            sVt[(c4 + 2) * VT_STRIDE + row] = f2bf(vv.z);
            sVt[(c4 + 3) * VT_STRIDE + row] = f2bf(vv.w);
        }
        // Prefetch next KV tile into L2 while we compute (global_prefetch_b8)
        if (kt + 1 < SEQ / KTILE) {
            __builtin_prefetch(Kp + (size_t)(kb + KTILE) * DHEAD + tid * 32, 0, 0);
            __builtin_prefetch(Vp + (size_t)(kb + KTILE) * DHEAD + tid * 32, 0, 0);
        }
        __syncthreads();

        // ---- S = Q @ K^T : 4 column-subtiles x 4 k-chunks of WMMA bf16 ----
        // B layout: lane = column j (half-waves split contraction), d-halves {8h..8h+7},{16+8h..16+8h+7}
        v8f s[4];
#pragma unroll
        for (int jt = 0; jt < 4; ++jt) {
            v8f c = vzero;
#pragma unroll
            for (int kc = 0; kc < 4; ++kc) {
                const int boff = (jt * 16 + ln16) * KT_STRIDE + kc * 32 + hi * 8;
                v16bf b = ldsfrag(sK, boff, boff + 16);
                c = __builtin_amdgcn_wmma_f32_16x16x32_bf16(
                        false, qf[kc], false, b, (short)0, c, false, false);
            }
            s[jt] = c;
        }

        // ---- Online softmax (row r lives in VGPR r of each fragment, cols across 16 lanes) ----
        float corr[8];
#pragma unroll
        for (int r = 0; r < 8; ++r) {
            float mx = fmaxf(fmaxf(s[0][r], s[1][r]), fmaxf(s[2][r], s[3][r]));
#pragma unroll
            for (int off = 8; off >= 1; off >>= 1) mx = fmaxf(mx, __shfl_xor(mx, off, 16));
            const float mnew = fmaxf(m[r], mx);
            corr[r] = __expf(m[r] - mnew);
            float sum = 0.0f;
#pragma unroll
            for (int jt = 0; jt < 4; ++jt) {
                const float p = __expf(s[jt][r] - mnew);
                s[jt][r] = p;
                sum += p;
            }
#pragma unroll
            for (int off = 8; off >= 1; off >>= 1) sum += __shfl_xor(sum, off, 16);
            l[r] = l[r] * corr[r] + sum;
            m[r] = mnew;
        }

        // ---- Rescale O accumulators ----
#pragma unroll
        for (int dt = 0; dt < 8; ++dt)
#pragma unroll
            for (int r = 0; r < 8; ++r)
                acc[dt][r] *= corr[r];

        // ---- Stage P (C-layout) -> LDS -> reload in A-layout as bf16 ----
        unsigned short* pw = sP + wave * 16 * PW_STRIDE;
#pragma unroll
        for (int jt = 0; jt < 4; ++jt)
#pragma unroll
            for (int r = 0; r < 8; ++r)
                pw[(r + hi * 8) * PW_STRIDE + jt * 16 + ln16] = f2bf(s[jt][r]);
        // (wave-private region; per-wave LDS ordering is in-order, compiler inserts s_wait_dscnt)

        // ---- O += P @ V : 2 j-chunks x 8 d-subtiles of WMMA bf16 ----
#pragma unroll
        for (int jc = 0; jc < 2; ++jc) {
            const int aoff = ln16 * PW_STRIDE + jc * 32 + hi * 8;
            v16bf a = ldsfrag(pw, aoff, aoff + 16);
#pragma unroll
            for (int dt = 0; dt < 8; ++dt) {
                const int boff = (dt * 16 + ln16) * VT_STRIDE + jc * 32 + hi * 8;
                v16bf b = ldsfrag(sVt, boff, boff + 16);
                acc[dt] = __builtin_amdgcn_wmma_f32_16x16x32_bf16(
                              false, a, false, b, (short)0, acc[dt], false, false);
            }
        }
    }

    // ---- Normalize and store (C-layout scatter: 2 x 64B contiguous runs per store) ----
#pragma unroll
    for (int r = 0; r < 8; ++r) {
        const float rl = 1.0f / l[r];
        float* op = Op + (size_t)(qb + wave * 16 + r + hi * 8) * DHEAD + ln16;
#pragma unroll
        for (int dt = 0; dt < 8; ++dt)
            op[dt * 16] = acc[dt][r] * rl;
    }
}

extern "C" void kernel_launch(void* const* d_in, const int* in_sizes, int n_in,
                              void* d_out, int out_size, void* d_ws, size_t ws_size,
                              hipStream_t stream) {
    (void)in_sizes; (void)n_in; (void)out_size; (void)d_ws; (void)ws_size;
    const float* q = (const float*)d_in[0];
    const float* k = (const float*)d_in[1];
    const float* v = (const float*)d_in[2];
    float* out = (float*)d_out;

    dim3 grid(SEQ / QTILE, 2 * 16);   // (q-tiles, B*H)
    dim3 block(256);                  // 8 waves of 32
    fa_fwd_bf16wmma<<<grid, block, 0, stream>>>(q, k, v, out);
}
